// Decoder_15281493639858
// MI455X (gfx1250) — matrix operand order repaired and checked by reading, not compiled, not run on hardware
//
#include <hip/hip_runtime.h>
#include <math.h>

#define HID 1024
#define SEQ 2048
#define VOCAB 50257

typedef __bf16 bf16_t;
typedef bf16_t bf16x16 __attribute__((ext_vector_type(16)));
typedef float  f32x8  __attribute__((ext_vector_type(8)));

__device__ __forceinline__ unsigned short f2bf(float f) {
    unsigned int u = __float_as_uint(f);
    u += 0x7fffu + ((u >> 16) & 1u);          // round-to-nearest-even
    return (unsigned short)(u >> 16);
}

// ---------- 1. enc (2048x1024 f32) -> bf16, row-major ----------
__global__ __launch_bounds__(256) void k_convert_enc(const float4* __restrict__ in,
                                                     ushort4* __restrict__ out) {
    int i = blockIdx.x * 256 + threadIdx.x;   // 524288 threads * 4 elems
    float4 v = in[i];
    ushort4 o; o.x = f2bf(v.x); o.y = f2bf(v.y); o.z = f2bf(v.z); o.w = f2bf(v.w);
    out[i] = o;
}

// ---------- 2. B = attn_W[1024:2048][:] -> Bt[n][k] bf16 (transpose) ----------
__global__ __launch_bounds__(256) void k_transpose_B(const float* __restrict__ attn_W,
                                                     unsigned short* __restrict__ Bt) {
    __shared__ float tile[32][33];
    int n0 = blockIdx.x * 32, k0 = blockIdx.y * 32;
    #pragma unroll
    for (int i = 0; i < 4; ++i) {
        int k = k0 + threadIdx.y + i * 8;
        tile[threadIdx.y + i * 8][threadIdx.x] =
            attn_W[(size_t)(HID + k) * HID + n0 + threadIdx.x];
    }
    __syncthreads();
    #pragma unroll
    for (int i = 0; i < 4; ++i) {
        int n = n0 + threadIdx.y + i * 8;
        Bt[(size_t)n * HID + k0 + threadIdx.x] = f2bf(tile[threadIdx.x][threadIdx.y + i * 8]);
    }
}

// ---------- 3/4. c[j] = sum_i h_last[i] * attn_W[i][j], split-i partials ----------
__global__ __launch_bounds__(256) void k_c_partial(const float* __restrict__ hid,
                                                   const float* __restrict__ attn_W,
                                                   float* __restrict__ c_part) {
    int j = (blockIdx.x & 3) * 256 + threadIdx.x;
    int ch = blockIdx.x >> 2;                 // 0..7, 128 i's each
    float s = 0.f;
    for (int i = ch * 128; i < ch * 128 + 128; ++i)
        s += hid[i] * attn_W[(size_t)i * HID + j];
    c_part[ch * HID + j] = s;
}
__global__ __launch_bounds__(256) void k_c_reduce(const float* __restrict__ c_part,
                                                  float* __restrict__ c) {
    int j = blockIdx.x * 256 + threadIdx.x;
    float s = 0.f;
    #pragma unroll
    for (int ch = 0; ch < 8; ++ch) s += c_part[ch * HID + j];
    c[j] = s;
}

// ---------- 5. WMMA: T = enc @ Bt^T ; scores_part = tanh(T + c) . v ----------
__global__ __launch_bounds__(256) void k_attn_scores_wmma(
        const unsigned short* __restrict__ encB,   // [2048][1024] bf16
        const unsigned short* __restrict__ BtB,    // [1024 n][1024 k] bf16
        const float* __restrict__ cvec,            // [1024]
        const float* __restrict__ vvec,            // [1024] attn_v
        float* __restrict__ part_scores)           // [8][2048]
{
    const int lane = threadIdx.x & 31;
    const int wave = threadIdx.x >> 5;
    const int W    = blockIdx.x * 8 + wave;        // 0..1023
    const int rowt = W >> 3;                       // 0..127 (S tile of 16)
    const int g    = W & 7;                        // j-group (128 columns)
    const int srow0 = rowt * 16;
    const int m  = lane & 15;
    const int hi = lane >> 4;

    const unsigned short* arow = encB + (size_t)(srow0 + m) * HID;

    f32x8 acc[8];
    f32x8 zero = {0.f,0.f,0.f,0.f,0.f,0.f,0.f,0.f};
    #pragma unroll
    for (int t = 0; t < 8; ++t) acc[t] = zero;

    for (int kk = 0; kk < 32; ++kk) {
        // A fragment (16x32 bf16): lane m holds K = kbase..+7 and kbase+16..+23
        union { uint4 u[2]; bf16x16 v; } A;
        const int kbase = kk * 32 + hi * 8;
        A.u[0] = *(const uint4*)(arow + kbase);
        A.u[1] = *(const uint4*)(arow + kbase + 16);
        #pragma unroll
        for (int t = 0; t < 8; ++t) {
            // B fragment (32x16): column n, K = kk*32 + hi*8 .. +7 and +16..+23 in Bt
            const int n = (g * 8 + t) * 16 + m;
            union { uint4 u[2]; bf16x16 v; } B;
            const unsigned short* bp = BtB + (size_t)n * HID + kk * 32 + hi * 8;
            B.u[0] = *(const uint4*)(bp);
            B.u[1] = *(const uint4*)(bp + 16);
            acc[t] = __builtin_amdgcn_wmma_f32_16x16x32_bf16(
                         false, A.v, false, B.v, (short)0, acc[t], false, false);
        }
    }

    // epilogue: lane holds rows r+hi*8, column n_local = m
    float part[8];
    #pragma unroll
    for (int r = 0; r < 8; ++r) part[r] = 0.f;
    #pragma unroll
    for (int t = 0; t < 8; ++t) {
        const int j = (g * 8 + t) * 16 + m;
        const float cj = cvec[j], vj = vvec[j];
        #pragma unroll
        for (int r = 0; r < 8; ++r)
            part[r] += tanhf(acc[t][r] + cj) * vj;
    }
    // reduce over the 16 columns (lanes within each half)
    #pragma unroll
    for (int r = 0; r < 8; ++r) {
        float v = part[r];
        v += __shfl_xor(v, 1, 32);
        v += __shfl_xor(v, 2, 32);
        v += __shfl_xor(v, 4, 32);
        v += __shfl_xor(v, 8, 32);
        if (m == 0) part_scores[g * SEQ + srow0 + hi * 8 + r] = v;
    }
}

// ---------- 6. combine j-group partials ----------
__global__ __launch_bounds__(256) void k_scores_reduce(const float* __restrict__ psc,
                                                       float* __restrict__ scores) {
    int s = blockIdx.x * 256 + threadIdx.x;
    float v = 0.f;
    #pragma unroll
    for (int gg = 0; gg < 8; ++gg) v += psc[gg * SEQ + s];
    scores[s] = v;
}

// ---------- 7. softmax over 2048 scores (single block, deterministic) ----------
__global__ __launch_bounds__(256) void k_softmax_scores(const float* __restrict__ scores,
                                                        float* __restrict__ w) {
    __shared__ float sm[256];
    int tid = threadIdx.x;
    float m = -1e30f;
    for (int i = tid; i < SEQ; i += 256) m = fmaxf(m, scores[i]);
    sm[tid] = m; __syncthreads();
    for (int s = 128; s > 0; s >>= 1) { if (tid < s) sm[tid] = fmaxf(sm[tid], sm[tid+s]); __syncthreads(); }
    m = sm[0]; __syncthreads();
    float sum = 0.f;
    for (int i = tid; i < SEQ; i += 256) sum += expf(scores[i] - m);
    sm[tid] = sum; __syncthreads();
    for (int s = 128; s > 0; s >>= 1) { if (tid < s) sm[tid] += sm[tid+s]; __syncthreads(); }
    float inv = 1.0f / sm[0];
    for (int i = tid; i < SEQ; i += 256) w[i] = expf(scores[i] - m) * inv;
}

// ---------- 8/9. context = w @ enc, split-S ----------
__global__ __launch_bounds__(256) void k_context_partial(const float* __restrict__ w,
                                                         const float* __restrict__ enc,
                                                         float* __restrict__ part) {
    int j  = (blockIdx.x & 3) * 256 + threadIdx.x;
    int ch = blockIdx.x >> 2;                  // 0..15, 128 rows each
    float s = 0.f;
    for (int r = ch * 128; r < ch * 128 + 128; ++r)
        s += w[r] * enc[(size_t)r * HID + j];
    part[ch * HID + j] = s;
}
__global__ __launch_bounds__(256) void k_context_reduce(const float* __restrict__ part,
                                                        float* __restrict__ ctx) {
    int j = blockIdx.x * 256 + threadIdx.x;
    float s = 0.f;
    #pragma unroll
    for (int ch = 0; ch < 16; ++ch) s += part[ch * HID + j];
    ctx[j] = s;
}

// ---------- 10. GRU GEMVs: gi = W_ih@[emb;ctx]+b_ih ; gh = W_hh@h+b_hh ----------
__global__ __launch_bounds__(256) void k_gru_gemv(const int* __restrict__ token,
        const float* __restrict__ emb, const float* __restrict__ ctx,
        const float* __restrict__ hid,
        const float* __restrict__ W_ih, const float* __restrict__ b_ih,
        const float* __restrict__ W_hh, const float* __restrict__ b_hh,
        float* __restrict__ gi, float* __restrict__ gh) {
    int lane = threadIdx.x & 31, wave = threadIdx.x >> 5;
    int row = blockIdx.x * 8 + wave;           // 0..6143
    int tok = token[0];
    float s = 0.f;
    if (row < 3 * HID) {
        const float* Wr = W_ih + (size_t)row * (2 * HID);
        for (int it = 0; it < 64; ++it) {
            int i = it * 32 + lane;
            float xv = (i < HID) ? emb[(size_t)tok * HID + i] : ctx[i - HID];
            s += Wr[i] * xv;
        }
        for (int msk = 16; msk; msk >>= 1) s += __shfl_xor(s, msk, 32);
        if (lane == 0) gi[row] = s + b_ih[row];
    } else {
        int o = row - 3 * HID;
        const float* Wr = W_hh + (size_t)o * HID;
        for (int it = 0; it < 32; ++it) { int i = it * 32 + lane; s += Wr[i] * hid[i]; }
        for (int msk = 16; msk; msk >>= 1) s += __shfl_xor(s, msk, 32);
        if (lane == 0) gh[o] = s + b_hh[o];
    }
}

// ---------- 11. GRU elementwise ----------
__global__ __launch_bounds__(256) void k_gru_elem(const float* __restrict__ gi,
        const float* __restrict__ gh, const float* __restrict__ hid,
        float* __restrict__ h_new, float* __restrict__ out_tail) {
    int k = blockIdx.x * 256 + threadIdx.x;
    float r = 1.f / (1.f + expf(-(gi[k] + gh[k])));
    float z = 1.f / (1.f + expf(-(gi[HID + k] + gh[HID + k])));
    float n = tanhf(gi[2 * HID + k] + r * gh[2 * HID + k]);
    float h = (1.f - z) * n + z * hid[k];
    h_new[k] = h; out_tail[k] = h;
}

// ---------- 12. logits = out_W @ h_new + out_b (206 MB stream, BW-bound) ----------
__global__ __launch_bounds__(256) void k_logits(const float4* __restrict__ out_W,
        const float* __restrict__ out_b, const float4* __restrict__ h4,
        float* __restrict__ logits) {
    int lane = threadIdx.x & 31, wave = threadIdx.x >> 5;
    int row = blockIdx.x * 8 + wave;
    if (row >= VOCAB) return;
    const float4* Wr = out_W + (size_t)row * (HID / 4);
    float s = 0.f;
    #pragma unroll
    for (int it = 0; it < 8; ++it) {
        float4 a = Wr[it * 32 + lane];
        float4 h = h4[it * 32 + lane];
        s += a.x * h.x + a.y * h.y + a.z * h.z + a.w * h.w;
    }
    for (int msk = 16; msk; msk >>= 1) s += __shfl_xor(s, msk, 32);
    if (lane == 0) logits[row] = s + out_b[row];
}

// ---------- 13/14/15. log-softmax over 50257 (online combine, deterministic) ----------
__global__ __launch_bounds__(256) void k_lse_partial(const float* __restrict__ logits,
                                                     float* __restrict__ part) {
    __shared__ float sm[256];
    int tid = threadIdx.x;
    float m = -1e30f;
    for (int i = blockIdx.x * 256 + tid; i < VOCAB; i += 128 * 256) m = fmaxf(m, logits[i]);
    sm[tid] = m; __syncthreads();
    for (int s = 128; s > 0; s >>= 1) { if (tid < s) sm[tid] = fmaxf(sm[tid], sm[tid+s]); __syncthreads(); }
    float bm = sm[0]; __syncthreads();
    float sum = 0.f;
    for (int i = blockIdx.x * 256 + tid; i < VOCAB; i += 128 * 256) sum += expf(logits[i] - bm);
    sm[tid] = sum; __syncthreads();
    for (int s = 128; s > 0; s >>= 1) { if (tid < s) sm[tid] += sm[tid+s]; __syncthreads(); }
    if (tid == 0) { part[blockIdx.x * 2] = bm; part[blockIdx.x * 2 + 1] = sm[0]; }
}
__global__ __launch_bounds__(128) void k_lse_final(const float* __restrict__ part,
                                                   float* __restrict__ lse) {
    __shared__ float sm[128];
    int tid = threadIdx.x;
    float bm = part[tid * 2], bs = part[tid * 2 + 1];
    sm[tid] = bm; __syncthreads();
    for (int s = 64; s > 0; s >>= 1) { if (tid < s) sm[tid] = fmaxf(sm[tid], sm[tid+s]); __syncthreads(); }
    float m = sm[0]; __syncthreads();
    sm[tid] = bs * expf(bm - m); __syncthreads();
    for (int s = 64; s > 0; s >>= 1) { if (tid < s) sm[tid] += sm[tid+s]; __syncthreads(); }
    if (tid == 0) lse[0] = m + logf(sm[0]);
}
__global__ __launch_bounds__(256) void k_write_out(const float* __restrict__ logits,
        const float* __restrict__ lse, float* __restrict__ out) {
    int i = blockIdx.x * 256 + threadIdx.x;
    if (i < VOCAB) out[i] = logits[i] - lse[0];
}

extern "C" void kernel_launch(void* const* d_in, const int* in_sizes, int n_in,
                              void* d_out, int out_size, void* d_ws, size_t ws_size,
                              hipStream_t stream) {
    (void)in_sizes; (void)n_in; (void)out_size; (void)ws_size;
    const int*   token  = (const int*)  d_in[0];
    const float* hidden = (const float*)d_in[1];
    const float* enc    = (const float*)d_in[2];
    const float* emb    = (const float*)d_in[3];
    const float* attn_W = (const float*)d_in[4];
    const float* attn_v = (const float*)d_in[5];
    const float* W_ih   = (const float*)d_in[6];
    const float* b_ih   = (const float*)d_in[7];
    const float* W_hh   = (const float*)d_in[8];
    const float* b_hh   = (const float*)d_in[9];
    const float* out_W  = (const float*)d_in[10];
    const float* out_b  = (const float*)d_in[11];
    float* out = (float*)d_out;
    char*  ws  = (char*)d_ws;

    unsigned short* encB = (unsigned short*)(ws + 0);         // 4 MB
    unsigned short* BtB  = (unsigned short*)(ws + 4194304);   // 2 MB
    float* c_part  = (float*)(ws + 6291456);
    float* cvec    = (float*)(ws + 6324224);
    float* psc     = (float*)(ws + 6328320);
    float* scores  = (float*)(ws + 6393856);
    float* wbuf    = (float*)(ws + 6402048);
    float* cxp     = (float*)(ws + 6410240);
    float* context = (float*)(ws + 6475776);
    float* gi      = (float*)(ws + 6479872);
    float* gh      = (float*)(ws + 6492160);
    float* h_new   = (float*)(ws + 6504448);
    float* logits  = (float*)(ws + 6508544);
    float* lsep    = (float*)(ws + 6709760);
    float* lse     = (float*)(ws + 6710784);

    k_convert_enc     <<<2048, 256, 0, stream>>>((const float4*)enc, (ushort4*)encB);
    k_transpose_B     <<<dim3(32, 32), dim3(32, 8), 0, stream>>>(attn_W, BtB);
    k_c_partial       <<<32, 256, 0, stream>>>(hidden, attn_W, c_part);
    k_c_reduce        <<<4, 256, 0, stream>>>(c_part, cvec);
    k_attn_scores_wmma<<<128, 256, 0, stream>>>(encB, BtB, cvec, attn_v, psc);
    k_scores_reduce   <<<8, 256, 0, stream>>>(psc, scores);
    k_softmax_scores  <<<1, 256, 0, stream>>>(scores, wbuf);
    k_context_partial <<<64, 256, 0, stream>>>(wbuf, enc, cxp);
    k_context_reduce  <<<4, 256, 0, stream>>>(cxp, context);
    k_gru_gemv        <<<768, 256, 0, stream>>>(token, emb, context, hidden,
                                                W_ih, b_ih, W_hh, b_hh, gi, gh);
    k_gru_elem        <<<4, 256, 0, stream>>>(gi, gh, hidden, h_new, out + VOCAB);
    k_logits          <<<(VOCAB + 7) / 8, 256, 0, stream>>>((const float4*)out_W, out_b,
                                                            (const float4*)h_new, logits);
    k_lse_partial     <<<128, 256, 0, stream>>>(logits, lsep);
    k_lse_final       <<<1, 128, 0, stream>>>(lsep, lse);
    k_write_out       <<<(VOCAB + 255) / 256, 256, 0, stream>>>(logits, lse, out);
}
